// SPEncoder_33105607918151
// MI455X (gfx1250) — matrix operand order repaired and checked by reading, not compiled, not run on hardware
//
#include <hip/hip_runtime.h>

// ---------------------------------------------------------------------------
// SPEncoder forward for MI455X (gfx1250, wave32, WMMA + async LDS staging).
// ---------------------------------------------------------------------------

#define SLEN 2048
#define BSZ 4
#define DIM 512
#define NHEAD 8
#define DHEAD 64
#define FDIM 2048
#define NROWS (SLEN * BSZ)          // 8192
#define EMBED_SCALE 22.6274169979695f  // sqrt(512)

typedef __bf16 bf16;
typedef __attribute__((ext_vector_type(16))) __bf16 bf16x16;
typedef __attribute__((ext_vector_type(8)))  float  f32x8;

#if defined(__gfx1250__) && __has_builtin(__builtin_amdgcn_global_load_async_to_lds_b128) && __has_builtin(__builtin_amdgcn_s_wait_asynccnt)
#define USE_ASYNC 1
#else
#define USE_ASYNC 0
#endif

__device__ __forceinline__ bf16 f2bf(float f) {
  unsigned int u = __builtin_bit_cast(unsigned int, f);
  unsigned int r = (u + 0x7FFFu + ((u >> 16) & 1u)) >> 16;   // round-to-nearest-even
  unsigned short s = (unsigned short)r;
  return __builtin_bit_cast(bf16, s);
}

#if USE_ASYNC
typedef int vi4 __attribute__((vector_size(16)));   // matches builtin's v4i pointee
#define ASG __attribute__((address_space(1)))
#define ASL __attribute__((address_space(3)))
__device__ __forceinline__ ASL vi4* to_lds(void* p) {
  return (ASL vi4*)(unsigned int)(unsigned long long)p;      // LDS offset = low 32 bits
}
__device__ __forceinline__ ASG vi4* to_glb(const void* p) {
  return (ASG vi4*)(unsigned long long)p;
}
#endif

// -------------------------- small elementwise kernels ----------------------

__global__ void cvt_bf16_kernel(const float* __restrict__ in, bf16* __restrict__ out, int n) {
  int i = blockIdx.x * blockDim.x + threadIdx.x;
  if (i < n) out[i] = f2bf(in[i]);
}

__global__ void embed_kernel(const float* __restrict__ x, float* __restrict__ out, int total) {
  int i = blockIdx.x * blockDim.x + threadIdx.x;
  if (i >= total) return;
  int d = i & (DIM - 1);
  int s = i >> 11;                       // / (BSZ*DIM) = 2048
  const float c = -9.21034037197618f / 255.0f;   // -ln(10000)/(half-1)
  float pos;
  if (d < 256) pos = __sinf((float)s * __expf(c * (float)d));
  else         pos = __cosf((float)s * __expf(c * (float)(d - 256)));
  out[i] = x[i] * EMBED_SCALE + pos;
}

__global__ void bcast_hidden_kernel(const float* __restrict__ init, float* __restrict__ h,
                                    int stream_idx, int total) {
  int i = blockIdx.x * blockDim.x + threadIdx.x;
  if (i < total) h[i] = init[stream_idx * DIM + (i & (DIM - 1))];
}

__global__ void add_kernel(const float* __restrict__ a, const float* __restrict__ b,
                           float* __restrict__ out, int n) {
  int i = blockIdx.x * blockDim.x + threadIdx.x;
  if (i < n) out[i] = a[i] + b[i];
}

// -------------------------- LayerNorm (one wave per row, D=512) ------------

__global__ __launch_bounds__(256) void ln_kernel(const float* __restrict__ x,
                                                 const float* __restrict__ g,
                                                 const float* __restrict__ b,
                                                 float* __restrict__ outf,
                                                 bf16*  __restrict__ outbf,
                                                 int nrows) {
  int row  = blockIdx.x * 8 + (threadIdx.x >> 5);
  int lane = threadIdx.x & 31;
  if (row >= nrows) return;
  const float* xr = x + (size_t)row * DIM;
  float vals[16];
  float s = 0.f, ss = 0.f;
#pragma unroll
  for (int i = 0; i < 16; ++i) {
    float v = xr[lane + 32 * i];
    vals[i] = v; s += v; ss += v * v;
  }
#pragma unroll
  for (int o = 16; o >= 1; o >>= 1) {
    s  += __shfl_xor(s, o, 32);
    ss += __shfl_xor(ss, o, 32);
  }
  float mean = s * (1.f / DIM);
  float var  = ss * (1.f / DIM) - mean * mean;
  float rstd = rsqrtf(var + 1e-5f);
#pragma unroll
  for (int i = 0; i < 16; ++i) {
    int d = lane + 32 * i;
    float v = (vals[i] - mean) * rstd * g[d] + b[d];
    if (outf)  outf [(size_t)row * DIM + d] = v;
    if (outbf) outbf[(size_t)row * DIM + d] = f2bf(v);
  }
}

// -------------------------- windowed attention (PF=3) ----------------------

__global__ __launch_bounds__(256) void attn_kernel(const float* __restrict__ Q,
                                                   const float* __restrict__ K,
                                                   const float* __restrict__ V,
                                                   bf16* __restrict__ O) {
  int idx = blockIdx.x * blockDim.x + threadIdx.x;     // (s*BS+b)*NH + n
  if (idx >= SLEN * BSZ * NHEAD) return;
  int n  = idx & (NHEAD - 1);
  int sb = idx >> 3;
  int b  = sb & (BSZ - 1);
  int s  = sb >> 2;
  const float* q = Q + (size_t)sb * DIM + n * DHEAD;
  int sp[3];
#pragma unroll
  for (int p = 0; p < 3; ++p) { int s2 = s - 2 + p; if (s2 < 0) s2 += SLEN; sp[p] = s2; }
  float logit[3];
#pragma unroll
  for (int p = 0; p < 3; ++p) {
    const float* k = K + ((size_t)(sp[p] * BSZ + b)) * DIM + n * DHEAD;
    float acc = 0.f;
#pragma unroll 8
    for (int d = 0; d < DHEAD; ++d) acc = fmaf(q[d], k[d], acc);
    logit[p] = acc;
  }
  float m  = fmaxf(logit[0], fmaxf(logit[1], logit[2]));
  float e0 = __expf(logit[0] - m), e1 = __expf(logit[1] - m), e2 = __expf(logit[2] - m);
  float inv = 1.f / (e0 + e1 + e2);
  const float* v0 = V + ((size_t)(sp[0] * BSZ + b)) * DIM + n * DHEAD;
  const float* v1 = V + ((size_t)(sp[1] * BSZ + b)) * DIM + n * DHEAD;
  const float* v2 = V + ((size_t)(sp[2] * BSZ + b)) * DIM + n * DHEAD;
  bf16* o = O + (size_t)sb * DIM + n * DHEAD;
#pragma unroll 8
  for (int d = 0; d < DHEAD; ++d)
    o[d] = f2bf((e0 * v0[d] + e1 * v1[d] + e2 * v2[d]) * inv);
}

// -------------------------- WMMA bf16 GEMM ---------------------------------
// C[M,N] = epilogue( A[M,K](bf16) @ W[N,K]^T(bf16) )
// Block 256 thr (8 waves); block tile 128x128, BK=32; wave tile 32x64 (2x4 WMMA).
// Double-buffered LDS; staging via GLOBAL_LOAD_ASYNC_TO_LDS_B128 when available.

#define BM 128
#define BN 128
#define BK 32
#define LDST 40     // padded LDS row stride (elements); 16B-aligned chunk offsets

__device__ __forceinline__ bf16x16 load_a_frag(const bf16* As, int row0, int lane) {
  // ISA 16-bit A 16x32 layout: lanes 0-15 K={0..7,16..23}; lanes 16-31 K={8..15,24..31}
  int m  = lane & 15;
  int hi = lane >> 4;
  bf16x16 f;
#pragma unroll
  for (int v = 0; v < 8; ++v) {
    int kb = ((v & 4) ? 16 : 0) + (hi ? 8 : 0) + ((v & 3) << 1);
    f[2 * v]     = As[(row0 + m) * LDST + kb];
    f[2 * v + 1] = As[(row0 + m) * LDST + kb + 1];
  }
  return f;
}

__device__ __forceinline__ bf16x16 load_b_frag(const bf16* Ws, int col0, int lane) {
  // B is Kx16; B[k][n] = W[n][k]. lanes 0-15 K={0..7,16..23}; lanes 16-31 K={8..15,24..31}
  int n  = lane & 15;
  int hi = lane >> 4;
  bf16x16 f;
#pragma unroll
  for (int v = 0; v < 8; ++v) {
    int k = ((v & 4) ? 16 : 0) + (hi ? 8 : 0) + ((v & 3) << 1);
    f[2 * v]     = Ws[(col0 + n) * LDST + k];
    f[2 * v + 1] = Ws[(col0 + n) * LDST + k + 1];
  }
  return f;
}

template <int N, int K>
__global__ __launch_bounds__(256) void gemm_bf16_kernel(const bf16* __restrict__ A,
                                                        const bf16* __restrict__ W,
                                                        const float* __restrict__ bias,
                                                        const float* __restrict__ res,
                                                        float* __restrict__ Cf,
                                                        bf16*  __restrict__ Cbf,
                                                        float scale, int relu) {
  __shared__ bf16 As[2][BM * LDST];
  __shared__ bf16 Ws[2][BN * LDST];
  const int tid  = threadIdx.x;
  const int lane = tid & 31;
  const int wave = tid >> 5;
  const int wm   = wave >> 1;          // 0..3  (rows of 32)
  const int wn   = wave & 1;           // 0..1  (cols of 64)
  const int bm   = blockIdx.y * BM;
  const int bn   = blockIdx.x * BN;
  constexpr int NSTEP = K / BK;

  // per-thread staging slots: rows r and r+64, 16B chunk q (compile-time strides)
  const int r = tid >> 2;              // 0..63
  const int q = (tid & 3) << 3;        // 0,8,16,24 elements
  const bf16* gA0 = A + (size_t)(bm + r) * K + q;
  const bf16* gA1 = gA0 + (size_t)64 * K;
  const bf16* gW0 = W + (size_t)(bn + r) * K + q;
  const bf16* gW1 = gW0 + (size_t)64 * K;
  const int lA = r * LDST + q;
  const int lB = lA + 64 * LDST;

  f32x8 acc[2][4] = {};

#if USE_ASYNC
  // prologue: stage step 0 into buffer 0
  __builtin_amdgcn_global_load_async_to_lds_b128(to_glb(gA0), to_lds(&As[0][lA]), 0, 0);
  __builtin_amdgcn_global_load_async_to_lds_b128(to_glb(gA1), to_lds(&As[0][lB]), 0, 0);
  __builtin_amdgcn_global_load_async_to_lds_b128(to_glb(gW0), to_lds(&Ws[0][lA]), 0, 0);
  __builtin_amdgcn_global_load_async_to_lds_b128(to_glb(gW1), to_lds(&Ws[0][lB]), 0, 0);
#else
  int4 ra0 = *(const int4*)gA0;
  int4 ra1 = *(const int4*)gA1;
  int4 rw0 = *(const int4*)gW0;
  int4 rw1 = *(const int4*)gW1;
#endif

  for (int i = 0; i < NSTEP; ++i) {
    const int cur = i & 1;
#if USE_ASYNC
    __builtin_amdgcn_s_wait_asynccnt(0);
    __syncthreads();
    if (i + 1 < NSTEP) {
      const int nxt = cur ^ 1;
      const int ko = (i + 1) * BK;
      __builtin_amdgcn_global_load_async_to_lds_b128(to_glb(gA0 + ko), to_lds(&As[nxt][lA]), 0, 0);
      __builtin_amdgcn_global_load_async_to_lds_b128(to_glb(gA1 + ko), to_lds(&As[nxt][lB]), 0, 0);
      __builtin_amdgcn_global_load_async_to_lds_b128(to_glb(gW0 + ko), to_lds(&Ws[nxt][lA]), 0, 0);
      __builtin_amdgcn_global_load_async_to_lds_b128(to_glb(gW1 + ko), to_lds(&Ws[nxt][lB]), 0, 0);
    }
#else
    *(int4*)&As[cur][lA] = ra0;
    *(int4*)&As[cur][lB] = ra1;
    *(int4*)&Ws[cur][lA] = rw0;
    *(int4*)&Ws[cur][lB] = rw1;
    __syncthreads();
    if (i + 1 < NSTEP) {
      const int ko = (i + 1) * BK;
      ra0 = *(const int4*)(gA0 + ko);
      ra1 = *(const int4*)(gA1 + ko);
      rw0 = *(const int4*)(gW0 + ko);
      rw1 = *(const int4*)(gW1 + ko);
    }
#endif
    bf16x16 af0 = load_a_frag(&As[cur][0], wm * 32 + 0,  lane);
    bf16x16 af1 = load_a_frag(&As[cur][0], wm * 32 + 16, lane);
    bf16x16 bfr[4];
#pragma unroll
    for (int j = 0; j < 4; ++j) bfr[j] = load_b_frag(&Ws[cur][0], wn * 64 + j * 16, lane);
#pragma unroll
    for (int j = 0; j < 4; ++j) {
      acc[0][j] = __builtin_amdgcn_wmma_f32_16x16x32_bf16(false, af0, false, bfr[j], (short)0, acc[0][j], false, false);
      acc[1][j] = __builtin_amdgcn_wmma_f32_16x16x32_bf16(false, af1, false, bfr[j], (short)0, acc[1][j], false, false);
    }
  }

  // epilogue: C layout — VGPR r, lane l -> M = r + 8*(l>>4), N = l&15
  const int nin  = lane & 15;
  const int rsel = lane >> 4;
#pragma unroll
  for (int i = 0; i < 2; ++i) {
#pragma unroll
    for (int j = 0; j < 4; ++j) {
      int col = bn + wn * 64 + j * 16 + nin;
      float bv = bias ? bias[col] : 0.f;
#pragma unroll
      for (int rr = 0; rr < 8; ++rr) {
        int row = bm + wm * 32 + i * 16 + rr + 8 * rsel;
        float v = (acc[i][j][rr] + bv) * scale;
        if (relu) v = fmaxf(v, 0.f);
        if (res)  v += res[(size_t)row * N + col];
        if (Cf)   Cf [(size_t)row * N + col] = v;
        if (Cbf)  Cbf[(size_t)row * N + col] = f2bf(v);
      }
    }
  }
}

// -------------------------- host orchestration -----------------------------

static void gemm(hipStream_t st, const bf16* A, const bf16* W, const float* bias,
                 const float* res, float* Cf, bf16* Cbf, int N, int K,
                 float scale, int relu) {
  dim3 blk(256);
  if (N == 512 && K == 512)
    gemm_bf16_kernel<512, 512><<<dim3(4, NROWS / BM), blk, 0, st>>>(A, W, bias, res, Cf, Cbf, scale, relu);
  else if (N == 2048 && K == 512)
    gemm_bf16_kernel<2048, 512><<<dim3(16, NROWS / BM), blk, 0, st>>>(A, W, bias, res, Cf, Cbf, scale, relu);
  else  // N == 512, K == 2048
    gemm_bf16_kernel<512, 2048><<<dim3(4, NROWS / BM), blk, 0, st>>>(A, W, bias, res, Cf, Cbf, scale, relu);
}

struct Ctx {
  const float *a, *t, *v, *init_h;
  const float *wq, *bq, *wk, *bk, *wv, *bv, *wo, *bo;
  const float *ln_g, *ln_b, *lnkv_g, *lnkv_b;
  const float *w1, *b1, *w2, *b2, *fln_g, *fln_b, *out_g, *out_b;
  bf16 *wq_bf, *wk_bf, *wv_bf, *wo_bf, *w1_bf, *w2_bf;
  float *emb[3], *h[3], *cx[6];
  bf16 *xn_bf, *cn_bf, *O_bf, *fin_bf, *H_bf;
  float *Qb, *Kb, *Vb;
  hipStream_t st;
};

static void run_spf(const Ctx& c, int i, const float* x, const float* ctx, bool has_kv, float* out) {
  const int M = NROWS;
  ln_kernel<<<M / 8, 256, 0, c.st>>>(x, c.ln_g + i * DIM, c.ln_b + i * DIM, nullptr, c.xn_bf, M);
  const bf16* kvin = c.xn_bf;
  if (ctx && has_kv) {
    // context LN + K/V projection done ONCE per row (window gather commutes with row-wise ops)
    ln_kernel<<<M / 8, 256, 0, c.st>>>(ctx, c.lnkv_g + i * DIM, c.lnkv_b + i * DIM, nullptr, c.cn_bf, M);
    kvin = c.cn_bf;
  }
  const float qscale = 0.125f;  // DH^-0.5 folded into Q epilogue
  gemm(c.st, c.xn_bf, c.wq_bf + (size_t)i * DIM * DIM, c.bq + i * DIM, nullptr, c.Qb, nullptr, DIM, DIM, qscale, 0);
  gemm(c.st, kvin,    c.wk_bf + (size_t)i * DIM * DIM, c.bk + i * DIM, nullptr, c.Kb, nullptr, DIM, DIM, 1.f, 0);
  gemm(c.st, kvin,    c.wv_bf + (size_t)i * DIM * DIM, c.bv + i * DIM, nullptr, c.Vb, nullptr, DIM, DIM, 1.f, 0);
  attn_kernel<<<(SLEN * BSZ * NHEAD) / 256, 256, 0, c.st>>>(c.Qb, c.Kb, c.Vb, c.O_bf);
  gemm(c.st, c.O_bf, c.wo_bf + (size_t)i * DIM * DIM, c.bo + i * DIM, x, out, nullptr, DIM, DIM, 1.f, 0);
}

static void run_ffn(const Ctx& c, int i, float* x) {
  const int M = NROWS;
  ln_kernel<<<M / 8, 256, 0, c.st>>>(x, c.fln_g + i * DIM, c.fln_b + i * DIM, nullptr, c.fin_bf, M);
  gemm(c.st, c.fin_bf, c.w1_bf + (size_t)i * FDIM * DIM, c.b1 + i * FDIM, nullptr, nullptr, c.H_bf, FDIM, DIM, 1.f, 1);
  gemm(c.st, c.H_bf,   c.w2_bf + (size_t)i * DIM * FDIM, c.b2 + i * DIM,  x, x, nullptr, DIM, FDIM, 1.f, 0);
}

extern "C" void kernel_launch(void* const* d_in, const int* in_sizes, int n_in,
                              void* d_out, int out_size, void* d_ws, size_t ws_size,
                              hipStream_t stream) {
  (void)in_sizes; (void)n_in; (void)out_size; (void)ws_size;
  Ctx c{};
  c.st = stream;
  c.a = (const float*)d_in[0];  c.t = (const float*)d_in[1];  c.v = (const float*)d_in[2];
  c.init_h = (const float*)d_in[3];
  c.wq = (const float*)d_in[4];  c.bq = (const float*)d_in[5];
  c.wk = (const float*)d_in[6];  c.bk = (const float*)d_in[7];
  c.wv = (const float*)d_in[8];  c.bv = (const float*)d_in[9];
  c.wo = (const float*)d_in[10]; c.bo = (const float*)d_in[11];
  c.ln_g = (const float*)d_in[12];  c.ln_b = (const float*)d_in[13];
  c.lnkv_g = (const float*)d_in[14]; c.lnkv_b = (const float*)d_in[15];
  c.w1 = (const float*)d_in[16]; c.b1 = (const float*)d_in[17];
  c.w2 = (const float*)d_in[18]; c.b2 = (const float*)d_in[19];
  c.fln_g = (const float*)d_in[20]; c.fln_b = (const float*)d_in[21];
  c.out_g = (const float*)d_in[22]; c.out_b = (const float*)d_in[23];

  size_t off = 0;
  char* ws = (char*)d_ws;
  auto carve = [&](size_t bytes) -> void* {
    void* p = ws + off;
    off = (off + bytes + 255) & ~(size_t)255;
    return p;
  };
  const size_t NW  = 9ull * DIM * DIM;
  const size_t NWF = 9ull * FDIM * DIM;
  const size_t MD  = (size_t)NROWS * DIM;
  c.wq_bf = (bf16*)carve(NW * 2);  c.wk_bf = (bf16*)carve(NW * 2);
  c.wv_bf = (bf16*)carve(NW * 2);  c.wo_bf = (bf16*)carve(NW * 2);
  c.w1_bf = (bf16*)carve(NWF * 2); c.w2_bf = (bf16*)carve(NWF * 2);
  for (int k = 0; k < 3; ++k) c.emb[k] = (float*)carve(MD * 4);
  for (int k = 0; k < 3; ++k) c.h[k]   = (float*)carve(MD * 4);
  for (int k = 0; k < 6; ++k) c.cx[k]  = (float*)carve(MD * 4);
  c.xn_bf  = (bf16*)carve(MD * 2);
  c.cn_bf  = (bf16*)carve(MD * 2);
  c.O_bf   = (bf16*)carve(MD * 2);
  c.fin_bf = (bf16*)carve(MD * 2);
  c.H_bf   = (bf16*)carve((size_t)NROWS * FDIM * 2);
  c.Qb = (float*)carve(MD * 4);
  c.Kb = (float*)carve(MD * 4);
  c.Vb = (float*)carve(MD * 4);

  auto cvt = [&](const float* src, bf16* dst, size_t n) {
    cvt_bf16_kernel<<<(unsigned)((n + 255) / 256), 256, 0, stream>>>(src, dst, (int)n);
  };
  cvt(c.wq, c.wq_bf, NW); cvt(c.wk, c.wk_bf, NW);
  cvt(c.wv, c.wv_bf, NW); cvt(c.wo, c.wo_bf, NW);
  cvt(c.w1, c.w1_bf, NWF); cvt(c.w2, c.w2_bf, NWF);

  const int total = (int)MD;
  embed_kernel<<<total / 256, 256, 0, stream>>>(c.a, c.emb[0], total);
  embed_kernel<<<total / 256, 256, 0, stream>>>(c.t, c.emb[1], total);
  embed_kernel<<<total / 256, 256, 0, stream>>>(c.v, c.emb[2], total);
  for (int k = 0; k < 3; ++k)
    bcast_hidden_kernel<<<total / 256, 256, 0, stream>>>(c.init_h, c.h[k], k, total);

  // hidden layer
  for (int k = 0; k < 3; ++k) {
    run_spf(c, k, c.h[k], c.emb[k], true, c.h[k]);
    run_ffn(c, k, c.h[k]);
  }
  // cross layer (six branches on frozen h's)
  run_spf(c, 3, c.h[0], c.h[1], true, c.cx[0]); run_ffn(c, 3, c.cx[0]);   // h_at
  run_spf(c, 4, c.h[1], c.h[2], true, c.cx[1]); run_ffn(c, 4, c.cx[1]);   // h_tv
  run_spf(c, 5, c.h[2], c.h[0], true, c.cx[2]); run_ffn(c, 5, c.cx[2]);   // h_va
  run_spf(c, 3, c.h[1], c.h[0], true, c.cx[3]); run_ffn(c, 3, c.cx[3]);   // h_ta
  run_spf(c, 4, c.h[2], c.h[1], true, c.cx[4]); run_ffn(c, 4, c.cx[4]);   // h_vt
  run_spf(c, 5, c.h[0], c.h[2], true, c.cx[5]); run_ffn(c, 5, c.cx[5]);   // h_av
  add_kernel<<<total / 256, 256, 0, stream>>>(c.cx[0], c.cx[5], c.h[0], total);
  add_kernel<<<total / 256, 256, 0, stream>>>(c.cx[3], c.cx[1], c.h[1], total);
  add_kernel<<<total / 256, 256, 0, stream>>>(c.cx[2], c.cx[4], c.h[2], total);
  // self layer
  for (int k = 0; k < 3; ++k) {
    run_spf(c, 6 + k, c.h[k], nullptr, false, c.h[k]);
    run_ffn(c, 6 + k, c.h[k]);
  }

  // final LN -> d_out (3, SL, BS, D) fp32
  float* out = (float*)d_out;
  for (int k = 0; k < 3; ++k)
    ln_kernel<<<NROWS / 8, 256, 0, stream>>>(c.h[k], c.out_g, c.out_b,
                                             out + (size_t)k * MD, nullptr, NROWS);
}
